// MambaBlock_70849780515030
// MI455X (gfx1250) — matrix operand order repaired and checked by reading, not compiled, not run on hardware
//
#include <hip/hip_runtime.h>

#define DEPTH   8
#define DIM     512
#define D_INNER 1024
#define D_STATE 16
#define D_CONV  4
#define DT_RANK 32
#define NB      2
#define SEQ     256            // H*W
#define ROWS    (NB * SEQ)     // 512 token rows

typedef __attribute__((ext_vector_type(16))) _Float16 v16h;
typedef __attribute__((ext_vector_type(8)))  _Float16 v8h;
typedef __attribute__((ext_vector_type(8)))  float    v8f;

// ---------------------------------------------------------------------------
// WMMA fragment loaders (f16, 16x16x32)
// A (16xK slice): dense 16-bit A layout: lane r holds row r; K chunks
//   [kk + half*8 .. +7] in v0-3 and [kk + 16 + half*8 .. +7] in v4-7.
// B (Kx16 slice, given as row-major Bt[n][k]): lane n holds column n;
//   K chunks [kk + half*8 .. +7] in v0-3 and [kk + 16 + half*8 .. +7] in v4-7.
// ---------------------------------------------------------------------------
__device__ __forceinline__ v16h load_a_frag(const _Float16* p) {
  union { v16h v; v8h h[2]; } u;
  u.h[0] = *(const v8h*)(p);
  u.h[1] = *(const v8h*)(p + 16);
  return u.v;
}
__device__ __forceinline__ v16h load_b_frag(const _Float16* p) {
  union { v16h v; v8h h[2]; } u;
  u.h[0] = *(const v8h*)(p);
  u.h[1] = *(const v8h*)(p + 16);
  return u.v;
}

// ---------------------------------------------------------------------------
// C[M,N] = A[M,K] * Bt[N,K]^T   (f16 in, f32 out). One wave -> 16(M) x 64(N).
// Grid sized exactly: waves = (M/16)*(N/64).
// ---------------------------------------------------------------------------
__global__ __launch_bounds__(256)
void gemm_nt_wmma(const _Float16* __restrict__ A, const _Float16* __restrict__ Bt,
                  float* __restrict__ C, int N, int K, int nSuper) {
  int tid  = blockIdx.x * 256 + threadIdx.x;
  int wave = tid >> 5;
  int lane = threadIdx.x & 31;
  int mt   = wave / nSuper;
  int nt   = wave - mt * nSuper;
  int half = lane >> 4;
  int r    = lane & 15;

  const _Float16* arow = A  + (size_t)(mt * 16 + r) * K;
  const _Float16* b0   = Bt + (size_t)(nt * 64 +  0 + r) * K;
  const _Float16* b1   = Bt + (size_t)(nt * 64 + 16 + r) * K;
  const _Float16* b2   = Bt + (size_t)(nt * 64 + 32 + r) * K;
  const _Float16* b3   = Bt + (size_t)(nt * 64 + 48 + r) * K;

  v8f acc0 = {}, acc1 = {}, acc2 = {}, acc3 = {};
  const int aoff = half * 8;
  const int boff = half * 8;

  for (int kk = 0; kk < K; kk += 32) {
    v16h a  = load_a_frag(arow + kk + aoff);
    v16h f0 = load_b_frag(b0 + kk + boff);
    v16h f1 = load_b_frag(b1 + kk + boff);
    v16h f2 = load_b_frag(b2 + kk + boff);
    v16h f3 = load_b_frag(b3 + kk + boff);
    acc0 = __builtin_amdgcn_wmma_f32_16x16x32_f16(false, a, false, f0, (short)0, acc0, false, false);
    acc1 = __builtin_amdgcn_wmma_f32_16x16x32_f16(false, a, false, f1, (short)0, acc1, false, false);
    acc2 = __builtin_amdgcn_wmma_f32_16x16x32_f16(false, a, false, f2, (short)0, acc2, false, false);
    acc3 = __builtin_amdgcn_wmma_f32_16x16x32_f16(false, a, false, f3, (short)0, acc3, false, false);
  }

  int mbase = mt * 16 + half * 8;
  int nbase = nt * 64 + r;
#pragma unroll
  for (int j = 0; j < 8; ++j) {
    float* crow = C + (size_t)(mbase + j) * N + nbase;
    crow[0]  = acc0[j];
    crow[16] = acc1[j];
    crow[32] = acc2[j];
    crow[48] = acc3[j];
  }
}

// ---------------------------------------------------------------------------
// Residual accumulate + RMSNorm, emits f16 normalized row for the GEMM.
// One block (256 threads) per row of 512.
// ---------------------------------------------------------------------------
__global__ __launch_bounds__(256)
void resnorm_kernel(const float* __restrict__ h, float* __restrict__ res,
                    _Float16* __restrict__ hn16, const float* __restrict__ w,
                    int first) {
  int row = blockIdx.x;
  int t   = threadIdx.x;
  const float* hrow = h   + (size_t)row * DIM;
  float*       rrow = res + (size_t)row * DIM;

  float r0 = hrow[t], r1 = hrow[t + 256];
  if (!first) { r0 += rrow[t]; r1 += rrow[t + 256]; }
  rrow[t] = r0; rrow[t + 256] = r1;

  float ss = r0 * r0 + r1 * r1;
#pragma unroll
  for (int off = 16; off > 0; off >>= 1) ss += __shfl_xor(ss, off, 32);
  __shared__ float wsum[8];
  if ((t & 31) == 0) wsum[t >> 5] = ss;
  __syncthreads();
  float total = 0.f;
#pragma unroll
  for (int i = 0; i < 8; ++i) total += wsum[i];

  float scale = rsqrtf(total * (1.0f / DIM) + 1e-5f);
  hn16[(size_t)row * DIM + t]       = (_Float16)(r0 * scale * w[t]);
  hn16[(size_t)row * DIM + t + 256] = (_Float16)(r1 * scale * w[t + 256]);
}

// f32 -> f16 conversion (weights)
__global__ __launch_bounds__(256)
void to_f16_kernel(const float* __restrict__ in, _Float16* __restrict__ out) {
  int i = blockIdx.x * 256 + threadIdx.x;
  out[i] = (_Float16)in[i];
}

// Depthwise causal conv (width 4, left pad 3) + bias + SiLU.
// xz: (ROWS, 2048) -> first 1024 cols are pre-conv x ; output xc: (ROWS, 1024)
__global__ __launch_bounds__(256)
void conv_silu_kernel(const float* __restrict__ xz, const float* __restrict__ cw,
                      const float* __restrict__ cb, float* __restrict__ xc) {
  int idx = blockIdx.x * 256 + threadIdx.x;   // ROWS * D_INNER
  int d = idx & (D_INNER - 1);
  int m = idx >> 10;
  int l = m & (SEQ - 1);
  float acc = cb[d];
#pragma unroll
  for (int j = 0; j < D_CONV; ++j) {
    int lj = l - 3 + j;
    if (lj >= 0) acc += cw[d * D_CONV + j] * xz[(size_t)(m - 3 + j) * (2 * D_INNER) + d];
  }
  xc[(size_t)m * D_INNER + d] = acc / (1.f + __expf(-acc));
}

// dbl[m,k] = sum_d xc[m,d] * Wx[k,d]   (k < 64). One block (64 thr) per row.
__global__ __launch_bounds__(64)
void xproj_kernel(const float* __restrict__ xc, const float* __restrict__ Wx,
                  float* __restrict__ dbl) {
  __shared__ float row[D_INNER];
  int m = blockIdx.x;
  int t = threadIdx.x;
  for (int i = t; i < D_INNER; i += 64) row[i] = xc[(size_t)m * D_INNER + i];
  __syncthreads();
  const float* wr = Wx + (size_t)t * D_INNER;
  float acc = 0.f;
  for (int d = 0; d < D_INNER; ++d) acc += row[d] * wr[d];
  dbl[m * 64 + t] = acc;
}

// dt[m,d] = softplus( sum_r dbl[m,r]*Wdt[d,r] + bdt[d] )
__global__ __launch_bounds__(256)
void dtproj_kernel(const float* __restrict__ dbl, const float* __restrict__ Wdt,
                   const float* __restrict__ bdt, float* __restrict__ dt) {
  int idx = blockIdx.x * 256 + threadIdx.x;   // ROWS * D_INNER
  int d = idx & (D_INNER - 1);
  int m = idx >> 10;
  const float* dr = dbl + m * 64;
  const float* wr = Wdt + (size_t)d * DT_RANK;
  float acc = bdt[d];
#pragma unroll
  for (int r = 0; r < DT_RANK; ++r) acc += dr[r] * wr[r];
  float sp = (acc > 20.f) ? acc : __logf(1.f + __expf(acc));
  dt[(size_t)m * D_INNER + d] = sp;
}

// Selective scan: one lane per (b, d, n) state element; 16 lanes per channel
// reduce y via shfl_xor tree each step.
__global__ __launch_bounds__(256)
void scan_kernel(const float* __restrict__ dt, const float* __restrict__ xc,
                 const float* __restrict__ dbl, const float* __restrict__ A_log,
                 float* __restrict__ ys) {
  int idx = blockIdx.x * 256 + threadIdx.x;   // NB * D_INNER * D_STATE = 32768
  int n = idx & (D_STATE - 1);
  int d = (idx >> 4) & (D_INNER - 1);
  int b = idx >> 14;
  float A = -__expf(A_log[d * D_STATE + n]);
  float s = 0.f;
  const int base = b * SEQ;
  for (int l = 0; l < SEQ; ++l) {
    int m = base + l;
    float dtv = dt[(size_t)m * D_INNER + d];
    float xv  = xc[(size_t)m * D_INNER + d];
    const float* bc = dbl + m * 64;
    float dA = __expf(dtv * A);
    s = dA * s + (dtv * xv) * bc[32 + n];
    float part = s * bc[48 + n];
    part += __shfl_xor(part, 1, 32);
    part += __shfl_xor(part, 2, 32);
    part += __shfl_xor(part, 4, 32);
    part += __shfl_xor(part, 8, 32);
    if (n == 0) ys[(size_t)m * D_INNER + d] = part;
  }
}

// y = (ys + D*xc) * silu(z) ; emit f16 for out_proj GEMM.
__global__ __launch_bounds__(256)
void gate_kernel(const float* __restrict__ ys, const float* __restrict__ xc,
                 const float* __restrict__ xz, const float* __restrict__ Dv,
                 _Float16* __restrict__ y16) {
  int idx = blockIdx.x * 256 + threadIdx.x;   // ROWS * D_INNER
  int d = idx & (D_INNER - 1);
  int m = idx >> 10;
  float z  = xz[(size_t)m * (2 * D_INNER) + D_INNER + d];
  float sz = z / (1.f + __expf(-z));
  float y  = (ys[(size_t)m * D_INNER + d] + Dv[d] * xc[(size_t)m * D_INNER + d]) * sz;
  y16[(size_t)m * D_INNER + d] = (_Float16)y;
}

// (B, C, L) -> (B, L, C)
__global__ __launch_bounds__(256)
void unpack_in_kernel(const float* __restrict__ x, float* __restrict__ h) {
  int idx = blockIdx.x * 256 + threadIdx.x;   // NB * DIM * SEQ
  int l = idx & (SEQ - 1);
  int c = (idx >> 8) & (DIM - 1);
  int b = idx >> 17;
  h[(size_t)(b * SEQ + l) * DIM + c] = x[idx];
}

// (B, L, C) -> (B, C, L)
__global__ __launch_bounds__(256)
void pack_out_kernel(const float* __restrict__ res, float* __restrict__ out) {
  int idx = blockIdx.x * 256 + threadIdx.x;   // NB * DIM * SEQ
  int l = idx & (SEQ - 1);
  int c = (idx >> 8) & (DIM - 1);
  int b = idx >> 17;
  out[idx] = res[(size_t)(b * SEQ + l) * DIM + c];
}

// ---------------------------------------------------------------------------
extern "C" void kernel_launch(void* const* d_in, const int* in_sizes, int n_in,
                              void* d_out, int out_size, void* d_ws, size_t ws_size,
                              hipStream_t stream) {
  (void)in_sizes; (void)n_in; (void)out_size; (void)ws_size;
  const float* x       = (const float*)d_in[0];
  const float* in_w    = (const float*)d_in[1];   // (8, 2048, 512)
  const float* conv_w  = (const float*)d_in[2];   // (8, 1024, 4)
  const float* conv_b  = (const float*)d_in[3];   // (8, 1024)
  const float* xproj_w = (const float*)d_in[4];   // (8, 64, 1024)
  const float* dt_w    = (const float*)d_in[5];   // (8, 1024, 32)
  const float* dt_b    = (const float*)d_in[6];   // (8, 1024)
  const float* A_log   = (const float*)d_in[7];   // (8, 1024, 16)
  const float* Dv      = (const float*)d_in[8];   // (8, 1024)
  const float* out_w   = (const float*)d_in[9];   // (8, 512, 1024)
  const float* norm_w  = (const float*)d_in[10];  // (8, 512)
  float* out = (float*)d_out;

  char* ws = (char*)d_ws;
  size_t off = 0;
  auto carve = [&](size_t bytes) -> void* {
    void* p = ws + off;
    off += (bytes + 255) & ~(size_t)255;
    return p;
  };
  float*    h    = (float*)   carve((size_t)ROWS * DIM * 4);            // 1 MB
  float*    res  = (float*)   carve((size_t)ROWS * DIM * 4);            // 1 MB
  float*    xz   = (float*)   carve((size_t)ROWS * 2 * D_INNER * 4);    // 4 MB
  float*    xc   = (float*)   carve((size_t)ROWS * D_INNER * 4);        // 2 MB
  float*    dbl  = (float*)   carve((size_t)ROWS * 64 * 4);             // 128 KB
  float*    dt   = (float*)   carve((size_t)ROWS * D_INNER * 4);        // 2 MB
  float*    ys   = (float*)   carve((size_t)ROWS * D_INNER * 4);        // 2 MB
  _Float16* hn16 = (_Float16*)carve((size_t)ROWS * DIM * 2);            // 0.5 MB
  _Float16* wi16 = (_Float16*)carve((size_t)2 * D_INNER * DIM * 2);     // 2 MB
  _Float16* wo16 = (_Float16*)carve((size_t)DIM * D_INNER * 2);         // 1 MB
  _Float16* y16  = (_Float16*)carve((size_t)ROWS * D_INNER * 2);        // 1 MB

  const int NE  = ROWS * D_INNER;        // 524288
  const int NT  = NB * DIM * SEQ;        // 262144

  unpack_in_kernel<<<NT / 256, 256, 0, stream>>>(x, h);

  for (int i = 0; i < DEPTH; ++i) {
    const float* Wi  = in_w    + (size_t)i * 2 * D_INNER * DIM;
    const float* cw  = conv_w  + (size_t)i * D_INNER * D_CONV;
    const float* cb  = conv_b  + (size_t)i * D_INNER;
    const float* Wx  = xproj_w + (size_t)i * 64 * D_INNER;
    const float* Wdt = dt_w    + (size_t)i * D_INNER * DT_RANK;
    const float* bdt = dt_b    + (size_t)i * D_INNER;
    const float* Al  = A_log   + (size_t)i * D_INNER * D_STATE;
    const float* Dl  = Dv      + (size_t)i * D_INNER;
    const float* Wo  = out_w   + (size_t)i * DIM * D_INNER;
    const float* nw  = norm_w  + (size_t)i * DIM;

    // residual += h ; hn16 = rmsnorm(residual) in f16
    resnorm_kernel<<<ROWS, 256, 0, stream>>>(h, res, hn16, nw, i == 0);

    // in_proj: xz[512,2048] = hn16[512,512] x Wi[2048,512]^T
    to_f16_kernel<<<(2 * D_INNER * DIM) / 256, 256, 0, stream>>>(Wi, wi16);
    gemm_nt_wmma<<<(ROWS / 16) * (2 * D_INNER / 64) * 32 / 256, 256, 0, stream>>>(
        hn16, wi16, xz, 2 * D_INNER, DIM, 2 * D_INNER / 64);

    // depthwise conv + silu
    conv_silu_kernel<<<NE / 256, 256, 0, stream>>>(xz, cw, cb, xc);

    // x_proj (dt-rank + B + C), dt_proj + softplus
    xproj_kernel<<<ROWS, 64, 0, stream>>>(xc, Wx, dbl);
    dtproj_kernel<<<NE / 256, 256, 0, stream>>>(dbl, Wdt, bdt, dt);

    // selective scan
    scan_kernel<<<(NB * D_INNER * D_STATE) / 256, 256, 0, stream>>>(dt, xc, dbl, Al, ys);

    // skip + gate -> f16
    gate_kernel<<<NE / 256, 256, 0, stream>>>(ys, xc, xz, Dl, y16);

    // out_proj: h[512,512] = y16[512,1024] x Wo[512,1024]^T
    to_f16_kernel<<<(DIM * D_INNER) / 256, 256, 0, stream>>>(Wo, wo16);
    gemm_nt_wmma<<<(ROWS / 16) * (DIM / 64) * 32 / 256, 256, 0, stream>>>(
        y16, wo16, h, DIM, D_INNER, DIM / 64);
  }

  // final output is the residual stream (final_norm_w unused in reference)
  pack_out_kernel<<<NT / 256, 256, 0, stream>>>(res, out);
}
